// Tacotron2_13984413515896
// MI455X (gfx1250) — matrix operand order repaired and checked by reading, not compiled, not run on hardware
//
#include <hip/hip_runtime.h>
#include <hip/hip_bf16.h>
#include <math.h>

// ---------------- model constants ----------------
#define BB   32
#define SS   256
#define NS   (BB*SS)      // 8192 encoder columns
#define TT   200
#define NT   (BB*TT)      // 6400 postnet columns
#define EMBD 256
#define MELD 80

typedef __attribute__((ext_vector_type(16))) _Float16 v16h;
typedef __attribute__((ext_vector_type(8)))  float    v8f;

// ---------------- fast transcendentals (inline, no libm call -> no spills) ----
__device__ __forceinline__ float fexp(float x) { return __expf(x); }
__device__ __forceinline__ float ftanh(float x) {
  x = fminf(fmaxf(x, -15.f), 15.f);
  float e = __expf(2.f * x);
  return (e - 1.f) / (e + 1.f);
}
__device__ __forceinline__ float sigf(float x) { return 1.f / (1.f + __expf(-x)); }

// ---------------- async global->LDS (CDNA5, ASYNCcnt tracked) ----------------
__device__ __forceinline__ void async_g2l_b128(void* lds, const void* g) {
  unsigned l = (unsigned)(uintptr_t)lds;
  unsigned long long a = (unsigned long long)(uintptr_t)g;
  asm volatile("global_load_async_to_lds_b128 %0, %1, off"
               :: "v"(l), "v"(a) : "memory");
}
__device__ __forceinline__ void wait_async0() {
  asm volatile("s_wait_asynccnt 0x0" ::: "memory");
}

// ---------------- WMMA helpers (CDNA5 16x16x32 f16 -> f32) ----------------
// A 16x32 f16: lane (0-15|16-31) row M=lane&15; k = half*16 + e (contiguous 16-run)
// B 32x16 f16 stored transposed [N][K]: lane col N=lane&15, k = half*16 + e (contiguous 16-run)
__device__ __forceinline__ v8f wmma16(v16h a, v16h b, v8f c) {
  return __builtin_amdgcn_wmma_f32_16x16x32_f16(false, a, false, b, (short)0, c,
                                                false, false);
}
__device__ __forceinline__ v8f zero8() {
  v8f z;
#pragma unroll
  for (int e = 0; e < 8; ++e) z[e] = 0.f;
  return z;
}
__device__ __forceinline__ v16h fragA(const _Float16* A, int ldk, int m0, int k0, int lane) {
  v16h f; int half = lane >> 4;
  const _Float16* p = A + (size_t)(m0 + (lane & 15)) * ldk + k0 + half * 16;
#pragma unroll
  for (int e = 0; e < 8; ++e) f[e] = p[e];
#pragma unroll
  for (int e = 8; e < 16; ++e) f[e] = p[e];
  return f;
}
__device__ __forceinline__ v16h fragB(const _Float16* Bt /*[N][K]*/, int ldk, int n0,
                                      int k0, int lane) {
  v16h f; int half = lane >> 4;
  const _Float16* p = Bt + (size_t)(n0 + (lane & 15)) * ldk + k0 + half * 16;
#pragma unroll
  for (int e = 0; e < 16; ++e) f[e] = p[e];
  return f;
}

// ---------------- small utility kernels ----------------
__global__ void k_embed(const int* __restrict__ tok, const float* __restrict__ emb,
                        _Float16* __restrict__ out /* [NS][EMBD] */) {
  int n = blockIdx.x * 256 + threadIdx.x;
  int e = blockIdx.y;
  out[(size_t)n * EMBD + e] = (_Float16)emb[tok[n] * EMBD + e];
}

__global__ void k_pack(const float* __restrict__ src, _Float16* __restrict__ dst,
                       int M, int K, int Mp, int Kp) {
  int idx = blockIdx.x * 256 + threadIdx.x;
  if (idx >= Mp * Kp) return;
  int m = idx / Kp, k = idx % Kp;
  float v = (m < M && k < K) ? src[m * K + k] : 0.f;
  dst[idx] = (_Float16)v;
}

__global__ void k_fold_bn(const float* __restrict__ b, const float* __restrict__ g,
                          const float* __restrict__ be, const float* __restrict__ mn,
                          const float* __restrict__ vr,
                          float* __restrict__ scale, float* __restrict__ shift,
                          int C, int Cp) {
  int c = blockIdx.x * 256 + threadIdx.x;
  if (c >= Cp) return;
  if (c < C) {
    float s = g[c] * rsqrtf(vr[c] + 1e-5f);
    scale[c] = s;
    shift[c] = (b[c] - mn[c]) * s + be[c];
  } else { scale[c] = 0.f; shift[c] = 0.f; }
}

__global__ void k_addb(const float* __restrict__ a, const float* __restrict__ b,
                       float* __restrict__ d, int n) {
  int i = blockIdx.x * 256 + threadIdx.x;
  if (i < n) d[i] = a[i] + b[i];
}

// Wcat [2048][1120]: cols 0..591 = dec Wih (mel|ctx), 592..1103 = dec Whh (h), pad 0
__global__ void k_pack_dec(const float* __restrict__ Wih, const float* __restrict__ Whh,
                           _Float16* __restrict__ dst) {
  int idx = blockIdx.x * 256 + threadIdx.x;
  if (idx >= 2048 * 1120) return;
  int m = idx / 1120, k = idx % 1120;
  float v = 0.f;
  if (k < 592)       v = Wih[m * 592 + k];
  else if (k < 1104) v = Whh[m * 512 + (k - 592)];
  dst[idx] = (_Float16)v;
}

// Wlp [80][1120]: ctx part (Wl cols 512..1023) -> k 80..591, h part (cols 0..511) -> k 592..1103
__global__ void k_pack_wl(const float* __restrict__ Wl, _Float16* __restrict__ dst) {
  int idx = blockIdx.x * 256 + threadIdx.x;
  if (idx >= 80 * 1120) return;
  int m = idx / 1120, k = idx % 1120;
  float v = 0.f;
  if (k >= 80 && k < 592)        v = Wl[m * 1024 + 512 + (k - 80)];
  else if (k >= 592 && k < 1104) v = Wl[m * 1024 + (k - 592)];
  dst[idx] = (_Float16)v;
}

// im2col conv1d K=5 same-pad per sample. x [Ncols][C], col [Ncols][Kp]
__global__ void k_im2col(const _Float16* __restrict__ x, _Float16* __restrict__ col,
                         int C, int S, int KK, int Kp) {
  int n = blockIdx.x * 256 + threadIdx.x;
  int row = blockIdx.y;
  _Float16 v = (_Float16)0.f;
  if (row < C * KK) {
    int c = row / KK, k = row % KK;
    int s = n % S;
    int ss = s + k - (KK >> 1);
    if (ss >= 0 && ss < S) v = x[(size_t)(n + ss - s) * C + c];
  }
  col[(size_t)n * Kp + row] = v;
}

// ---------------- tiled WMMA GEMM: out[n][m] = act(scale[m]*(A@B)[m][n]+shift[m]) -
// A f16 [M][K] row-major weights, B f16 [N][K] sample-major activations.
// M%128==0, N%64==0, K%32==0. Tiles staged via async global->LDS b128.
template <int ACT, bool WF32, bool WF16>
__global__ void __launch_bounds__(256)
k_gemm(const _Float16* __restrict__ A, const _Float16* __restrict__ B,
       float* __restrict__ Cf /* [N][M] */, _Float16* __restrict__ Ch /* [N][M] */,
       const float* __restrict__ scale, const float* __restrict__ shift,
       int M, int N, int K) {
  __shared__ __align__(16) _Float16 As[128 * 32];   // [128 m][32 k]
  __shared__ __align__(16) _Float16 Bs[64 * 32];    // [64 n][32 k] (transposed tile)
  const int tid = threadIdx.x, lane = tid & 31, wave = tid >> 5;
  const int wm = (wave >> 1) * 32;  // 0,32,64,96
  const int wn = (wave & 1) * 32;   // 0,32
  const int m0 = blockIdx.y * 128, n0 = blockIdx.x * 64;
  v8f acc[2][2];
  for (int i = 0; i < 2; ++i) for (int j = 0; j < 2; ++j) acc[i][j] = zero8();

  const int ar0 = tid >> 1, aq0 = (tid & 1) << 4;        // A: 256 rows*2 halves-chunks? no:
  for (int k0 = 0; k0 < K; k0 += 32) {
    // A tile 128x32 halves = 512 16B chunks (2 per thread), async to LDS
    {
      int c = tid, r = c >> 2, q = (c & 3) << 3;
      async_g2l_b128(&As[r * 32 + q], &A[(size_t)(m0 + r) * K + k0 + q]);
      c += 256; r = c >> 2; q = (c & 3) << 3;
      async_g2l_b128(&As[r * 32 + q], &A[(size_t)(m0 + r) * K + k0 + q]);
    }
    // B tile 64x32 halves = 256 16B chunks (1 per thread), async to LDS
    {
      int n = tid >> 2, q = (tid & 3) << 3;
      async_g2l_b128(&Bs[n * 32 + q], &B[(size_t)(n0 + n) * K + k0 + q]);
    }
    __builtin_prefetch(&A[(size_t)(m0 + (tid >> 1)) * K + k0 + 32], 0, 1);
    wait_async0();
    __syncthreads();
    v16h a0 = fragA(As, 32, wm, 0, lane);
    v16h a1 = fragA(As, 32, wm + 16, 0, lane);
    v16h b0 = fragB(Bs, 32, wn, 0, lane);
    v16h b1 = fragB(Bs, 32, wn + 16, 0, lane);
    acc[0][0] = wmma16(a0, b0, acc[0][0]);
    acc[0][1] = wmma16(a0, b1, acc[0][1]);
    acc[1][0] = wmma16(a1, b0, acc[1][0]);
    acc[1][1] = wmma16(a1, b1, acc[1][1]);
    __syncthreads();
  }
  const int half = lane >> 4;
  for (int i = 0; i < 2; ++i)
    for (int j = 0; j < 2; ++j) {
      int mb = m0 + wm + i * 16 + half * 8;
      int nb = n0 + wn + j * 16 + (lane & 15);
#pragma unroll
      for (int r = 0; r < 8; ++r) {
        int m = mb + r;
        float v = acc[i][j][r];
        float sc = scale ? scale[m] : 1.f;
        float sh = shift ? shift[m] : 0.f;
        v = v * sc + sh;
        if (ACT == 1)      v = v > 0.f ? v : 0.f;
        else if (ACT == 2) v = ftanh(v);
        if (WF32) Cf[(size_t)nb * M + m] = v;
        if (WF16) Ch[(size_t)nb * M + m] = (_Float16)v;
      }
    }
}

// ---------------- LSTM recurrence: persistent single WG, LDS state ----------------
#define LSTM_SM (32*256*2 + 256*32*4 + 1024*32*4)
__global__ void __launch_bounds__(256)
k_lstm(const float* __restrict__ xg /* [NS][1024] preact+bias */,
       const _Float16* __restrict__ Whh /* [1024][256] f16 */,
       _Float16* __restrict__ encout /* [NS][512] */, int reverse, int rowoff) {
  extern __shared__ __align__(16) char sm[];
  _Float16* hb = (_Float16*)sm;                        // [32 n][256 k]
  float*    cb = (float*)(sm + 32 * 256 * 2);          // [256*32]
  float*    gb = (float*)(sm + 32 * 256 * 2 + 256 * 32 * 4);  // [1024 m][32 n]
  const int tid = threadIdx.x, lane = tid & 31, wave = tid >> 5;
  const int half = lane >> 4;
  for (int i = tid; i < 256 * 32; i += 256) { hb[i] = (_Float16)0.f; cb[i] = 0.f; }
  __syncthreads();
  for (int t = 0; t < SS; ++t) {
    int st = reverse ? (SS - 1 - t) : t;
    // g = xg[:, :, st] + Whh @ h   (M=1024, N=32, K=256)
    for (int j = 0; j < 8; ++j) {
      int mrow = (wave + 8 * j) * 16;
      for (int nt = 0; nt < 2; ++nt) {
        int nb = nt * 16 + (lane & 15);
        v8f acc;
#pragma unroll
        for (int r = 0; r < 8; ++r)
          acc[r] = xg[(size_t)(nb * SS + st) * 1024 + mrow + half * 8 + r];
        for (int k0 = 0; k0 < 256; k0 += 32) {
          v16h a = fragA(Whh, 256, mrow, k0, lane);
          v16h b = fragB(hb, 256, nt * 16, k0, lane);
          acc = wmma16(a, b, acc);
        }
#pragma unroll
        for (int r = 0; r < 8; ++r) gb[(mrow + half * 8 + r) * 32 + nb] = acc[r];
      }
    }
    __syncthreads();
    for (int rep = 0; rep < 32; ++rep) {
      int p = tid + 256 * rep;  // 8192 (j,b) pairs
      int j = p >> 5, b = p & 31;
      float ig = gb[j * 32 + b], fg = gb[(256 + j) * 32 + b];
      float gg = gb[(512 + j) * 32 + b], og = gb[(768 + j) * 32 + b];
      float c = cb[p];
      c = sigf(fg) * c + sigf(ig) * ftanh(gg);
      float h = sigf(og) * ftanh(c);
      cb[p] = c;
      hb[b * 256 + j] = (_Float16)h;
      encout[(size_t)(b * SS + st) * 512 + rowoff + j] = (_Float16)h;
    }
    __syncthreads();
  }
}

// ---------------- Decoder: persistent single WG, 200 autoregressive steps ----------------
#define DEC_SM (32*1120*2 + 512*32*4 + 128*32*4 + 2048*32*2)
__global__ void __launch_bounds__(256)
k_decoder(const _Float16* __restrict__ Wd16 /* [128][512] */, const float* __restrict__ bd,
          const float* __restrict__ encproj /* [NS][128] */, const float* __restrict__ vw,
          const float* __restrict__ bv, const _Float16* __restrict__ enc16 /* [NS][512] */,
          const _Float16* __restrict__ Wcat /* [2048][1120] */, const float* __restrict__ bcat,
          const _Float16* __restrict__ Wlp /* [80][1120] */, const float* __restrict__ bl,
          float* __restrict__ mel_out /* [B][80][T] */,
          _Float16* __restrict__ melT /* [NT][80] */) {
  extern __shared__ __align__(16) char sm[];
  _Float16* Bv   = (_Float16*)sm;                                 // [32 n][1120 k] mel|ctx|h|pad
  float*    cb   = (float*)(sm + 32 * 1120 * 2);                  // [512*32]
  float*    decs = (float*)(sm + 32 * 1120 * 2 + 512 * 32 * 4);   // [128 m][32 n]
  _Float16* gb   = (_Float16*)((char*)decs + 128 * 32 * 4);       // [2048 m][32 n]
  float*    scr  = (float*)gb;                                    // [32][256] aliased
  const int tid = threadIdx.x, lane = tid & 31, wave = tid >> 5;
  const int half = lane >> 4;
  for (int i = tid; i < 32 * 1120; i += 256) Bv[i] = (_Float16)0.f;
  for (int i = tid; i < 512 * 32; i += 256)  cb[i] = 0.f;
  __syncthreads();

  for (int t = 0; t < TT; ++t) {
    // (1) dec = Wd @ h + bd : M=128,N=32,K=512  (h = Bv cols 592..1103)
    for (int j = wave; j < 16; j += 8) {
      int mt = j >> 1, nt = j & 1;
      int nb = nt * 16 + (lane & 15);
      v8f acc = zero8();
      for (int k0 = 0; k0 < 512; k0 += 32) {
        v16h a = fragA(Wd16, 512, mt * 16, k0, lane);
        v16h b = fragB(Bv, 1120, nt * 16, 592 + k0, lane);
        acc = wmma16(a, b, acc);
      }
#pragma unroll
      for (int r = 0; r < 8; ++r) {
        int m = mt * 16 + half * 8 + r;
        decs[m * 32 + nb] = acc[r] + bd[m];
      }
    }
    __syncthreads();
    // (2) attention scores
    for (int rep = 0; rep < 32; ++rep) {
      int p = tid + 256 * rep;
      int b = p >> 8, s = p & 255;
      const float* ep = encproj + (size_t)(b * SS + s) * 128;
      float sum = bv[0];
      for (int a = 0; a < 128; ++a)
        sum += ftanh(ep[a] + decs[a * 32 + b]) * vw[a];
      scr[b * 256 + s] = sum;
    }
    __syncthreads();
    // (3) softmax over S per batch row
    if (tid < 32) {
      float mx = -1e30f;
      for (int s = 0; s < 256; ++s) mx = fmaxf(mx, scr[tid * 256 + s]);
      float sum = 0.f;
      for (int s = 0; s < 256; ++s) {
        float e = fexp(scr[tid * 256 + s] - mx);
        scr[tid * 256 + s] = e; sum += e;
      }
      float inv = 1.f / sum;
      for (int s = 0; s < 256; ++s) scr[tid * 256 + s] *= inv;
    }
    __syncthreads();
    // (4) ctx[d][b] -> Bv cols 80..591
    for (int rep = 0; rep < 64; ++rep) {
      int p = tid + 256 * rep;
      int d = p >> 5, b = p & 31;
      float a = 0.f;
      for (int s = 0; s < 256; ++s)
        a += scr[b * 256 + s] * (float)enc16[(size_t)(b * SS + s) * 512 + d];
      Bv[b * 1120 + 80 + d] = (_Float16)a;
    }
    __syncthreads();
    // (5) gates preact: Wcat @ [mel|ctx|h] + (bih+bhh) : M=2048,N=32,K=1120
    for (int i = 0; i < 32; ++i) {
      int j = wave * 32 + i;
      int mt = j >> 1, nt = j & 1;
      int nb = nt * 16 + (lane & 15);
      v8f acc = zero8();
      for (int k0 = 0; k0 < 1120; k0 += 32) {
        v16h a = fragA(Wcat, 1120, mt * 16, k0, lane);
        v16h b = fragB(Bv, 1120, nt * 16, k0, lane);
        acc = wmma16(a, b, acc);
      }
#pragma unroll
      for (int r = 0; r < 8; ++r) {
        int m = mt * 16 + half * 8 + r;
        gb[m * 32 + nb] = (_Float16)(acc[r] + bcat[m]);
      }
    }
    __syncthreads();
    // (6) LSTM cell update, h -> Bv cols 592..1103
    for (int rep = 0; rep < 64; ++rep) {
      int p = tid + 256 * rep;
      int j = p >> 5, b = p & 31;
      float ig = (float)gb[j * 32 + b],          fg = (float)gb[(512 + j) * 32 + b];
      float gg = (float)gb[(1024 + j) * 32 + b], og = (float)gb[(1536 + j) * 32 + b];
      float c = cb[p];
      c = sigf(fg) * c + sigf(ig) * ftanh(gg);
      float h = sigf(og) * ftanh(c);
      cb[p] = c;
      Bv[b * 1120 + 592 + j] = (_Float16)h;
    }
    __syncthreads();
    // (7) out = Wlp @ [.|ctx|h] + bl : M=80,N=32,K=1120 ; feed back as mel
    for (int j = wave; j < 10; j += 8) {
      int mt = j >> 1, nt = j & 1;
      int nb = nt * 16 + (lane & 15);
      v8f acc = zero8();
      for (int k0 = 0; k0 < 1120; k0 += 32) {
        v16h a = fragA(Wlp, 1120, mt * 16, k0, lane);
        v16h b = fragB(Bv, 1120, nt * 16, k0, lane);
        acc = wmma16(a, b, acc);
      }
#pragma unroll
      for (int r = 0; r < 8; ++r) {
        int m = mt * 16 + half * 8 + r;
        float v = acc[r] + bl[m];
        mel_out[(size_t)nb * MELD * TT + m * TT + t] = v;       // [B][80][T]
        melT[(size_t)(nb * TT + t) * MELD + m] = (_Float16)v;   // [NT][80] for postnet
        Bv[nb * 1120 + m] = (_Float16)v;                        // mel feedback
      }
    }
    __syncthreads();
  }
}

// ---------------- residual add: out2 = mel + postnet ----------------
__global__ void k_final(const float* __restrict__ mel, const float* __restrict__ postY,
                        float* __restrict__ out2) {
  int idx = blockIdx.x * 256 + threadIdx.x;
  if (idx >= BB * MELD * TT) return;
  int t = idx % TT;
  int m = (idx / TT) % MELD;
  int b = idx / (TT * MELD);
  out2[idx] = mel[idx] + postY[(size_t)(b * TT + t) * 128 + m];
}

// ---------------- host orchestration ----------------
extern "C" void kernel_launch(void* const* d_in, const int* in_sizes, int n_in,
                              void* d_out, int out_size, void* d_ws, size_t ws_size,
                              hipStream_t stream) {
  (void)in_sizes; (void)n_in; (void)out_size; (void)ws_size;
  const int*   text = (const int*)d_in[0];
  const float* emb  = (const float*)d_in[1];
  const float *cW[3], *cB[3], *cG[3], *cBe[3], *cM[3], *cV[3];
  for (int i = 0; i < 3; ++i) {
    int b0 = 2 + 6 * i;
    cW[i]  = (const float*)d_in[b0 + 0]; cB[i] = (const float*)d_in[b0 + 1];
    cG[i]  = (const float*)d_in[b0 + 2]; cBe[i] = (const float*)d_in[b0 + 3];
    cM[i]  = (const float*)d_in[b0 + 4]; cV[i] = (const float*)d_in[b0 + 5];
  }
  const float* lWih[2] = {(const float*)d_in[20], (const float*)d_in[24]};
  const float* lWhh[2] = {(const float*)d_in[21], (const float*)d_in[25]};
  const float* lbih[2] = {(const float*)d_in[22], (const float*)d_in[26]};
  const float* lbhh[2] = {(const float*)d_in[23], (const float*)d_in[27]};
  const float* We = (const float*)d_in[28]; const float* be = (const float*)d_in[29];
  const float* Wd = (const float*)d_in[30]; const float* bd = (const float*)d_in[31];
  const float* Vw = (const float*)d_in[32]; const float* bv = (const float*)d_in[33];
  const float* dWih = (const float*)d_in[34]; const float* dWhh = (const float*)d_in[35];
  const float* dbih = (const float*)d_in[36]; const float* dbhh = (const float*)d_in[37];
  const float* Wl = (const float*)d_in[38]; const float* bl = (const float*)d_in[39];
  // d_in[40..41] = gate_p (unused by reference outputs)
  const float *pW[5], *pB[5], *pG[5], *pBe[5], *pM[5], *pV[5];
  for (int i = 0; i < 5; ++i) {
    int b0 = 42 + 6 * i;
    pW[i] = (const float*)d_in[b0 + 0]; pB[i] = (const float*)d_in[b0 + 1];
    pG[i] = (const float*)d_in[b0 + 2]; pBe[i] = (const float*)d_in[b0 + 3];
    pM[i] = (const float*)d_in[b0 + 4]; pV[i] = (const float*)d_in[b0 + 5];
  }

  char* wsb = (char*)d_ws; size_t off = 0;
  auto alloc = [&](size_t bytes) -> char* {
    char* p = wsb + off; off += (bytes + 255) & ~(size_t)255; return p;
  };
  _Float16* act0 = (_Float16*)alloc((size_t)NS * 512 * 2);   // [NS][C]
  _Float16* act1 = (_Float16*)alloc((size_t)NS * 512 * 2);
  _Float16* colb = (_Float16*)alloc((size_t)NS * 2560 * 2);  // [NS][Kp]
  int convK[3] = {1280, 2560, 2560};
  _Float16* Ac[3]; float *scl[3], *shf[3];
  for (int i = 0; i < 3; ++i) {
    Ac[i]  = (_Float16*)alloc((size_t)512 * convK[i] * 2);
    scl[i] = (float*)alloc(512 * 4);
    shf[i] = (float*)alloc(512 * 4);
  }
  _Float16 *Wih16[2], *Whh16[2]; float* shl[2];
  for (int d = 0; d < 2; ++d) {
    Wih16[d] = (_Float16*)alloc((size_t)1024 * 512 * 2);
    Whh16[d] = (_Float16*)alloc((size_t)1024 * 256 * 2);
    shl[d]   = (float*)alloc(1024 * 4);
  }
  float*    xg    = (float*)alloc((size_t)NS * 1024 * 4);    // [NS][1024]
  _Float16* enc16 = (_Float16*)alloc((size_t)NS * 512 * 2);  // [NS][512]
  float*    eproj = (float*)alloc((size_t)NS * 128 * 4);     // [NS][128]
  _Float16* We16  = (_Float16*)alloc((size_t)128 * 512 * 2);
  _Float16* Wd16  = (_Float16*)alloc((size_t)128 * 512 * 2);
  _Float16* Wcat  = (_Float16*)alloc((size_t)2048 * 1120 * 2);
  float*    bcat  = (float*)alloc(2048 * 4);
  _Float16* Wlp   = (_Float16*)alloc((size_t)80 * 1120 * 2);
  _Float16* melT  = (_Float16*)alloc((size_t)NT * MELD * 2); // [NT][80]
  int pKs[5] = {416, 2560, 2560, 2560, 2560};
  int pMs[5] = {512, 512, 512, 512, 128};
  int pKr[5] = {400, 2560, 2560, 2560, 2560};
  int pMr[5] = {512, 512, 512, 512, 80};
  _Float16* Ap[5]; float *pscl[5], *pshf[5];
  for (int l = 0; l < 5; ++l) {
    Ap[l]   = (_Float16*)alloc((size_t)pMs[l] * pKs[l] * 2);
    pscl[l] = (float*)alloc(512 * 4);
    pshf[l] = (float*)alloc(512 * 4);
  }
  _Float16* pcol  = (_Float16*)alloc((size_t)NT * 2560 * 2);  // [NT][Kp]
  _Float16* pa0   = (_Float16*)alloc((size_t)NT * 512 * 2);   // [NT][512]
  _Float16* pa1   = (_Float16*)alloc((size_t)NT * 512 * 2);
  float*    postY = (float*)alloc((size_t)NT * 128 * 4);      // [NT][128]

  // ---- encoder: embedding ----
  k_embed<<<dim3(NS / 256, EMBD), 256, 0, stream>>>(text, emb, act0);
  // ---- pack conv weights / fold BN ----
  for (int i = 0; i < 3; ++i) {
    int n = 512 * convK[i];
    k_pack<<<(n + 255) / 256, 256, 0, stream>>>(cW[i], Ac[i], 512, convK[i], 512, convK[i]);
    k_fold_bn<<<2, 256, 0, stream>>>(cB[i], cG[i], cBe[i], cM[i], cV[i],
                                     scl[i], shf[i], 512, 512);
  }
  // ---- conv chain (im2col + WMMA GEMM + BN + ReLU) ----
  _Float16* cur = act0; _Float16* nxt = act1; int cin = EMBD;
  for (int i = 0; i < 3; ++i) {
    k_im2col<<<dim3(NS / 256, convK[i]), 256, 0, stream>>>(cur, colb, cin, SS, 5, convK[i]);
    k_gemm<1, false, true><<<dim3(NS / 64, 512 / 128), 256, 0, stream>>>(
        Ac[i], colb, nullptr, nxt, scl[i], shf[i], 512, NS, convK[i]);
    _Float16* t = cur; cur = nxt; nxt = t; cin = 512;
  }
  // ---- BiLSTM ----
  for (int d = 0; d < 2; ++d) {
    k_pack<<<(1024 * 512 + 255) / 256, 256, 0, stream>>>(lWih[d], Wih16[d], 1024, 512, 1024, 512);
    k_pack<<<(1024 * 256 + 255) / 256, 256, 0, stream>>>(lWhh[d], Whh16[d], 1024, 256, 1024, 256);
    k_addb<<<4, 256, 0, stream>>>(lbih[d], lbhh[d], shl[d], 1024);
  }
  for (int d = 0; d < 2; ++d) {
    k_gemm<0, true, false><<<dim3(NS / 64, 1024 / 128), 256, 0, stream>>>(
        Wih16[d], cur, xg, nullptr, nullptr, shl[d], 1024, NS, 512);
    k_lstm<<<1, 256, LSTM_SM, stream>>>(xg, Whh16[d], enc16, d, d * 256);
  }
  // ---- attention / decoder weight packs ----
  k_pack<<<(128 * 512 + 255) / 256, 256, 0, stream>>>(We, We16, 128, 512, 128, 512);
  k_pack<<<(128 * 512 + 255) / 256, 256, 0, stream>>>(Wd, Wd16, 128, 512, 128, 512);
  k_pack_dec<<<(2048 * 1120 + 255) / 256, 256, 0, stream>>>(dWih, dWhh, Wcat);
  k_addb<<<8, 256, 0, stream>>>(dbih, dbhh, bcat, 2048);
  k_pack_wl<<<(80 * 1120 + 255) / 256, 256, 0, stream>>>(Wl, Wlp);
  // enc_proj = We @ enc_out + be  (hoisted, constant per decoder step)
  k_gemm<0, true, false><<<dim3(NS / 64, 1), 256, 0, stream>>>(
      We16, enc16, eproj, nullptr, nullptr, be, 128, NS, 512);
  // ---- autoregressive decoder ----
  k_decoder<<<1, 256, DEC_SM, stream>>>(Wd16, bd, eproj, Vw, bv, enc16, Wcat, bcat,
                                        Wlp, bl, (float*)d_out, melT);
  // ---- postnet ----
  for (int l = 0; l < 5; ++l) {
    int n = pMs[l] * pKs[l];
    k_pack<<<(n + 255) / 256, 256, 0, stream>>>(pW[l], Ap[l], pMr[l], pKr[l], pMs[l], pKs[l]);
    k_fold_bn<<<(pMs[l] + 255) / 256, 256, 0, stream>>>(
        pB[l], pG[l], pBe[l], pM[l], pV[l], pscl[l], pshf[l], pMr[l], pMs[l]);
  }
  _Float16* pc = melT; _Float16* pn = pa0; int pcin = MELD;
  for (int l = 0; l < 5; ++l) {
    k_im2col<<<dim3(NT / 256, pKs[l]), 256, 0, stream>>>(pc, pcol, pcin, TT, 5, pKs[l]);
    if (l < 4) {
      k_gemm<2, false, true><<<dim3(NT / 64, pMs[l] / 128), 256, 0, stream>>>(
          Ap[l], pcol, nullptr, pn, pscl[l], pshf[l], pMs[l], NT, pKs[l]);
      pc = pn; pn = (pn == pa0) ? pa1 : pa0; pcin = 512;
    } else {
      k_gemm<0, true, false><<<dim3(NT / 64, 1), 256, 0, stream>>>(
          Ap[4], pcol, postY, nullptr, pscl[4], pshf[4], 128, NT, 2560);
    }
  }
  // ---- residual ----
  k_final<<<(BB * MELD * TT + 255) / 256, 256, 0, stream>>>(
      (const float*)d_out, postY, (float*)d_out + (size_t)BB * MELD * TT);
}
